// PostLNEncoder_14937896256194
// MI455X (gfx1250) — matrix operand order repaired and checked by reading, not compiled, not run on hardware
//
#include <hip/hip_runtime.h>
#include <hip/hip_bf16.h>
#include <math.h>

// MI455X / gfx1250, wave32. All GEMMs via V_WMMA_F32_16X16X32_BF16 (f32 accum),
// tile staging via GLOBAL_LOAD_ASYNC_TO_LDS_B128 (ASYNCcnt) with double buffering.
// B=4, S=2048, D=1024, H=16, HD=64, F=4096, M=B*S=8192.

typedef __attribute__((ext_vector_type(8)))  float  v8f;
typedef __attribute__((ext_vector_type(16))) __bf16 v16bf;
typedef __attribute__((ext_vector_type(8)))  __bf16 v8bf;

#define MB   8192   // rows of all activation GEMMs (B*S)
#define DD   1024
#define FF   4096
#define SS   2048
#define HH   16
#define HD   64

// ---------------------------------------------------------------------------
// CDNA5 async copy: 16 bytes global -> LDS, tracked by ASYNCcnt.
// LDS address operand is the per-lane LDS byte offset (generic ptr low 32 bits).
// ---------------------------------------------------------------------------
__device__ __forceinline__ void async_copy16(void* lds_ptr, const void* gptr) {
  unsigned int  loff = (unsigned int)(size_t)lds_ptr;
  unsigned long long ga = (unsigned long long)(size_t)gptr;
  asm volatile("global_load_async_to_lds_b128 %0, %1, off"
               :: "v"(loff), "v"(ga) : "memory");
}
__device__ __forceinline__ void wait_async0() {
  asm volatile("s_wait_asynccnt 0x0" ::: "memory");
}

// ---------------------------------------------------------------------------
// WMMA fragment loaders (bf16, 16x16x32).
// A layout (ISA 7.12.2, 16-bit A 16x32): lane<16 -> M=lane, K chunks {0..7,16..23};
//                                        lane>=16 -> M=lane-16, K chunks {8..15,24..31}.
// B layout (matches A): lane<16 -> N=lane, K chunks {0..7,16..23};
//                       lane>=16 -> N=lane-16, K chunks {8..15,24..31}.
// ---------------------------------------------------------------------------
__device__ __forceinline__ v16bf load_frag_a(const __bf16* base, int row0, int ld, int k0) {
  const int lane = threadIdx.x & 31;
  const int hi   = lane >> 4;
  const __bf16* p = base + (size_t)(row0 + (lane & 15)) * ld + k0;
  v8bf c0 = *(const v8bf*)(p + hi * 8);
  v8bf c1 = *(const v8bf*)(p + 16 + hi * 8);
  return __builtin_shufflevector(c0, c1, 0,1,2,3,4,5,6,7,8,9,10,11,12,13,14,15);
}

__device__ __forceinline__ v16bf load_frag_b(const __bf16* base, int col0, int ld, int k0) {
  const int lane = threadIdx.x & 31;
  const int hi   = lane >> 4;
  const __bf16* p = base + (size_t)(col0 + (lane & 15)) * ld + k0 + hi * 8;
  v8bf c0 = *(const v8bf*)(p);
  v8bf c1 = *(const v8bf*)(p + 16);
  return __builtin_shufflevector(c0, c1, 0,1,2,3,4,5,6,7,8,9,10,11,12,13,14,15);
}

__device__ __forceinline__ v8f wmma_bf16(v16bf a, v16bf b, v8f c) {
  return __builtin_amdgcn_wmma_f32_16x16x32_bf16(false, a, false, b, (short)0, c, false, false);
}

__device__ __forceinline__ float gelu_exact(float v) {
  return 0.5f * v * (1.0f + erff(v * 0.70710678118654752f));
}

// ---------------------------------------------------------------------------
// Tiled GEMM: C[M,N] = A[M,K](bf16) * B, with B given transposed BT[N,K](bf16).
// Block tile 128x128, 4 waves (2x2) of 64x64. K-step 32, double-buffered LDS
// filled by async-to-LDS DMA so the next tile streams in under the WMMAs.
// EPI 0: out_bf = bf16(acc + bias)
// EPI 1: out_f  = acc + bias + aux   (residual add)
// EPI 2: out_bf = bf16(gelu(acc + bias))
// ---------------------------------------------------------------------------
#define GLDT 40   // padded LDS stride (bf16 elems); 80B rows keep 16B chunk align

template <int EPI>
__global__ void __launch_bounds__(128)
gemm_bf16(const __bf16* __restrict__ A, const __bf16* __restrict__ BT,
          const float* __restrict__ bias, const float* __restrict__ aux,
          float* __restrict__ outF, __bf16* __restrict__ outBF,
          int M, int N, int K) {
  __shared__ __align__(16) __bf16 As[2][128 * GLDT];
  __shared__ __align__(16) __bf16 Bs[2][128 * GLDT];

  const int t  = threadIdx.x;
  const int w  = t >> 5;
  const int wm = (w >> 1) * 64;
  const int wn = (w & 1) * 64;
  const int bm = blockIdx.y * 128;
  const int bn = blockIdx.x * 128;

  v8f acc[4][4];
#pragma unroll
  for (int i = 0; i < 4; ++i)
#pragma unroll
    for (int j = 0; j < 4; ++j)
      acc[i][j] = (v8f){0.f,0.f,0.f,0.f,0.f,0.f,0.f,0.f};

  // Async-stage a 128x32 A tile and 128x32 BT tile into buffer `buf`.
  // 512 chunks of 8 bf16 per tile; 4 consecutive threads cover one row (64B).
  auto issue = [&](int k0, int buf) {
#pragma unroll
    for (int s = 0; s < 4; ++s) {
      int cidx = s * 128 + t;
      int r = cidx >> 2, c = (cidx & 3) * 8;
      async_copy16(&As[buf][r * GLDT + c], &A [(size_t)(bm + r) * K + k0 + c]);
      async_copy16(&Bs[buf][r * GLDT + c], &BT[(size_t)(bn + r) * K + k0 + c]);
    }
  };

  issue(0, 0);
  wait_async0();
  __syncthreads();

  const int nk = K >> 5;
  for (int it = 0; it < nk; ++it) {
    const int cur = it & 1;
    if (it + 1 < nk) issue((it + 1) << 5, cur ^ 1);   // DMA next tile under compute

    v16bf af[4], bfb[4];
#pragma unroll
    for (int fm = 0; fm < 4; ++fm) af[fm]  = load_frag_a(&As[cur][0], wm + fm * 16, GLDT, 0);
#pragma unroll
    for (int fn = 0; fn < 4; ++fn) bfb[fn] = load_frag_b(&Bs[cur][0], wn + fn * 16, GLDT, 0);
#pragma unroll
    for (int fm = 0; fm < 4; ++fm)
#pragma unroll
      for (int fn = 0; fn < 4; ++fn)
        acc[fm][fn] = wmma_bf16(af[fm], bfb[fn], acc[fm][fn]);

    wait_async0();     // next tile fully in LDS
    __syncthreads();   // all waves done reading `cur`
  }

  // Epilogue. C fragment: VGPR i -> row i (lanes 0-15) / i+8 (lanes 16-31),
  // column = lane&15.
  const int lane = t & 31;
  const int hi   = lane >> 4;
  const int cl   = lane & 15;
#pragma unroll
  for (int fm = 0; fm < 4; ++fm) {
#pragma unroll
    for (int fn = 0; fn < 4; ++fn) {
      const int col  = bn + wn + fn * 16 + cl;
      const float bc = bias[col];
#pragma unroll
      for (int i = 0; i < 8; ++i) {
        const int row = bm + wm + fm * 16 + i + 8 * hi;
        float v = acc[fm][fn][i] + bc;
        if (EPI == 1) {
          v += aux[(size_t)row * N + col];
          outF[(size_t)row * N + col] = v;
        } else if (EPI == 2) {
          outBF[(size_t)row * N + col] = (__bf16)gelu_exact(v);
        } else {
          outBF[(size_t)row * N + col] = (__bf16)v;
        }
      }
    }
  }
}

// ---------------------------------------------------------------------------
// Flash attention: one block per (64 q-rows, head, batch); 4 waves x 16 rows.
// K tiles stream in via async-to-LDS DMA; V is transposed on store so PV's
// B-fragment reads are contiguous. Online softmax, WMMA for QK^T and PV.
// q,k,v,out all bf16 in [B,S,D] layout with head h occupying cols h*64..h*64+63.
// ---------------------------------------------------------------------------
#define ALD 72

__global__ void __launch_bounds__(128)
flash_attn(const __bf16* __restrict__ Q, const __bf16* __restrict__ Kb,
           const __bf16* __restrict__ Vb, __bf16* __restrict__ O) {
  __shared__ __align__(16) __bf16 KT[64 * ALD];          // [key][hd]
  __shared__ __align__(16) __bf16 VT[64 * ALD];          // [hd][key] (transposed)
  __shared__ __align__(16) __bf16 PT[4 * 16 * ALD];      // per-wave P tiles

  const int t    = threadIdx.x;
  const int w    = t >> 5;
  const int lane = t & 31;
  const int hi   = lane >> 4;
  const int cl   = lane & 15;
  const int s0   = blockIdx.x * 64;
  const int h    = blockIdx.y;
  const int b    = blockIdx.z;
  const size_t basebh = ((size_t)b * SS) * DD + (size_t)h * HD;

  // Stage this block's Q rows into KT (async DMA), pull per-wave A fragments.
#pragma unroll
  for (int s = 0; s < 4; ++s) {
    int cidx = s * 128 + t;
    int r = cidx >> 3, c = (cidx & 7) * 8;
    async_copy16(&KT[r * ALD + c], &Q[basebh + (size_t)(s0 + r) * DD + c]);
  }
  wait_async0();
  __syncthreads();
  v16bf qa[2];
  qa[0] = load_frag_a(KT, w * 16, ALD, 0);
  qa[1] = load_frag_a(KT, w * 16, ALD, 32);
  __syncthreads();

  float m[8], l[8];
  v8f o[4];
#pragma unroll
  for (int i = 0; i < 8; ++i) { m[i] = -1e30f; l[i] = 0.f; }
#pragma unroll
  for (int j = 0; j < 4; ++j) o[j] = (v8f){0.f,0.f,0.f,0.f,0.f,0.f,0.f,0.f};

  __bf16* Pw = &PT[w * 16 * ALD];

  for (int t0 = 0; t0 < SS; t0 += 64) {
    // K tile via async DMA (straight); V tile transposed manually.
#pragma unroll
    for (int s = 0; s < 4; ++s) {
      int cidx = s * 128 + t;
      int r = cidx >> 3, c = (cidx & 7) * 8;
      async_copy16(&KT[r * ALD + c], &Kb[basebh + (size_t)(t0 + r) * DD + c]);
      v8bf vv = *(const v8bf*)&Vb[basebh + (size_t)(t0 + r) * DD + c];
#pragma unroll
      for (int e = 0; e < 8; ++e) VT[(c + e) * ALD + r] = vv[e];
    }
    wait_async0();
    __syncthreads();

    // Scores: S[16 x 64] = Q[16x64] * K_tile^T  (BT == KT directly).
    v8f sf[4];
#pragma unroll
    for (int j = 0; j < 4; ++j) {
      sf[j] = (v8f){0.f,0.f,0.f,0.f,0.f,0.f,0.f,0.f};
#pragma unroll
      for (int kk = 0; kk < 2; ++kk) {
        v16bf bfr = load_frag_b(KT, j * 16, ALD, kk * 32);
        sf[j] = wmma_bf16(qa[kk], bfr, sf[j]);
      }
#pragma unroll
      for (int i = 0; i < 8; ++i) sf[j][i] *= 0.125f;   // 1/sqrt(64)
    }

    // Online softmax per row (row = 16 lanes of a half-wave).
#pragma unroll
    for (int i = 0; i < 8; ++i) {
      float mx = -1e30f;
#pragma unroll
      for (int j = 0; j < 4; ++j) mx = fmaxf(mx, sf[j][i]);
#pragma unroll
      for (int off = 1; off < 16; off <<= 1) mx = fmaxf(mx, __shfl_xor(mx, off, 32));
      const float mn    = fmaxf(m[i], mx);
      const float alpha = __expf(m[i] - mn);
      float rs = 0.f;
#pragma unroll
      for (int j = 0; j < 4; ++j) {
        float p = __expf(sf[j][i] - mn);
        rs += p;
        Pw[(i + 8 * hi) * ALD + j * 16 + cl] = (__bf16)p;
      }
#pragma unroll
      for (int off = 1; off < 16; off <<= 1) rs += __shfl_xor(rs, off, 32);
      l[i] = l[i] * alpha + rs;
      m[i] = mn;
#pragma unroll
      for (int j = 0; j < 4; ++j) o[j][i] *= alpha;
    }

    // PV: O[16x64] += P[16x64] * V[64x64]  (BT == VT).
    v16bf pa0 = load_frag_a(Pw, 0, ALD, 0);
    v16bf pa1 = load_frag_a(Pw, 0, ALD, 32);
#pragma unroll
    for (int j = 0; j < 4; ++j) {
      v16bf b0 = load_frag_b(VT, j * 16, ALD, 0);
      o[j] = wmma_bf16(pa0, b0, o[j]);
      v16bf b1 = load_frag_b(VT, j * 16, ALD, 32);
      o[j] = wmma_bf16(pa1, b1, o[j]);
    }
    __syncthreads();
  }

  // Normalize and store (bf16) into attn buffer, head-concatenated layout.
#pragma unroll
  for (int j = 0; j < 4; ++j)
#pragma unroll
    for (int i = 0; i < 8; ++i) {
      const int row = s0 + w * 16 + i + 8 * hi;
      O[basebh + (size_t)row * DD + j * 16 + cl] = (__bf16)(o[j][i] / l[i]);
    }
}

// ---------------------------------------------------------------------------
// LayerNorm over D=1024: one row per 256-thread block (8 wave32).
// ---------------------------------------------------------------------------
__global__ void __launch_bounds__(256)
layernorm_k(const float* __restrict__ in, const float* __restrict__ gg,
            const float* __restrict__ bb, float* __restrict__ outF,
            __bf16* __restrict__ outBF, int writeBF) {
  __shared__ float redS[8], redQ[8];
  const int row = blockIdx.x, t = threadIdx.x;
  const float4 v = *(const float4*)&in[(size_t)row * DD + t * 4];
  float s = v.x + v.y + v.z + v.w;
  float q = v.x * v.x + v.y * v.y + v.z * v.z + v.w * v.w;
#pragma unroll
  for (int off = 16; off; off >>= 1) {
    s += __shfl_xor(s, off, 32);
    q += __shfl_xor(q, off, 32);
  }
  if ((t & 31) == 0) { redS[t >> 5] = s; redQ[t >> 5] = q; }
  __syncthreads();
  s = 0.f; q = 0.f;
#pragma unroll
  for (int i = 0; i < 8; ++i) { s += redS[i]; q += redQ[i]; }
  const float mean = s * (1.0f / 1024.0f);
  const float var  = q * (1.0f / 1024.0f) - mean * mean;
  const float rstd = rsqrtf(var + 1e-5f);
  const int c0 = t * 4;
  float y0 = (v.x - mean) * rstd * gg[c0 + 0] + bb[c0 + 0];
  float y1 = (v.y - mean) * rstd * gg[c0 + 1] + bb[c0 + 1];
  float y2 = (v.z - mean) * rstd * gg[c0 + 2] + bb[c0 + 2];
  float y3 = (v.w - mean) * rstd * gg[c0 + 3] + bb[c0 + 3];
  float4 y; y.x = y0; y.y = y1; y.z = y2; y.w = y3;
  *(float4*)&outF[(size_t)row * DD + c0] = y;
  if (writeBF) {
    outBF[(size_t)row * DD + c0 + 0] = (__bf16)y0;
    outBF[(size_t)row * DD + c0 + 1] = (__bf16)y1;
    outBF[(size_t)row * DD + c0 + 2] = (__bf16)y2;
    outBF[(size_t)row * DD + c0 + 3] = (__bf16)y3;
  }
}

// ---------------------------------------------------------------------------
// Conversion / transposition helpers (run once per call; weights are small).
// ---------------------------------------------------------------------------
__global__ void __launch_bounds__(256)
convert_bf16(const float* __restrict__ in, __bf16* __restrict__ out, long total) {
  long idx = (long)blockIdx.x * 256 + threadIdx.x;
  if (idx < total) out[idx] = (__bf16)in[idx];
}

// BT[n*K + k] = W[k*N + n]   (W is [K,N] row-major f32)
__global__ void __launch_bounds__(256)
transpose_convert(const float* __restrict__ W, __bf16* __restrict__ BT, int K, int N) {
  long idx = (long)blockIdx.x * 256 + threadIdx.x;
  if (idx >= (long)K * N) return;
  int n = (int)(idx / K), k = (int)(idx - (long)n * K);
  BT[idx] = (__bf16)W[(size_t)k * N + n];
}

// Head weights W[H,D,HD] -> BT[n*D + d] with n = h*64 + hd (flattened [D, H*HD]^T)
__global__ void __launch_bounds__(256)
transpose_convert_head(const float* __restrict__ W, __bf16* __restrict__ BT) {
  long idx = (long)blockIdx.x * 256 + threadIdx.x;   // n*1024 + d, total 1024*1024
  if (idx >= (long)DD * DD) return;
  int n = (int)(idx >> 10), d = (int)(idx & 1023);
  BT[idx] = (__bf16)W[(size_t)(n >> 6) * (DD * HD) + (size_t)d * HD + (n & 63)];
}

// ---------------------------------------------------------------------------
// Host orchestration.
// ---------------------------------------------------------------------------
extern "C" void kernel_launch(void* const* d_in, const int* in_sizes, int n_in,
                              void* d_out, int out_size, void* d_ws, size_t ws_size,
                              hipStream_t stream) {
  (void)in_sizes; (void)n_in; (void)out_size; (void)ws_size;
  const float* x    = (const float*)d_in[0];
  const float* Wq   = (const float*)d_in[1];
  const float* bq   = (const float*)d_in[2];
  const float* Wk   = (const float*)d_in[3];
  const float* bk   = (const float*)d_in[4];
  const float* Wv   = (const float*)d_in[5];
  const float* bv   = (const float*)d_in[6];
  const float* Wo   = (const float*)d_in[7];
  const float* bo   = (const float*)d_in[8];
  const float* ln1g = (const float*)d_in[9];
  const float* ln1b = (const float*)d_in[10];
  const float* ln2g = (const float*)d_in[11];
  const float* ln2b = (const float*)d_in[12];
  const float* W1   = (const float*)d_in[13];
  const float* b1   = (const float*)d_in[14];
  const float* W2   = (const float*)d_in[15];
  const float* b2   = (const float*)d_in[16];
  float* out = (float*)d_out;

  char* wp = (char*)d_ws;
  auto alloc = [&](size_t bytes) {
    char* p = wp;
    wp += (bytes + 255) & ~(size_t)255;
    return p;
  };
  const size_t MD = (size_t)MB * DD;   // 8M elems
  const size_t MF = (size_t)MB * FF;   // 32M elems

  __bf16* xbf    = (__bf16*)alloc(MD * 2);
  __bf16* WqT    = (__bf16*)alloc((size_t)DD * DD * 2);
  __bf16* WkT    = (__bf16*)alloc((size_t)DD * DD * 2);
  __bf16* WvT    = (__bf16*)alloc((size_t)DD * DD * 2);
  __bf16* WoT    = (__bf16*)alloc((size_t)DD * DD * 2);
  __bf16* W1T    = (__bf16*)alloc((size_t)DD * FF * 2);
  __bf16* W2T    = (__bf16*)alloc((size_t)FF * DD * 2);
  __bf16* qbf    = (__bf16*)alloc(MD * 2);
  __bf16* kbf    = (__bf16*)alloc(MD * 2);
  __bf16* vbf    = (__bf16*)alloc(MD * 2);
  __bf16* attnbf = (__bf16*)alloc(MD * 2);
  float*  resid  = (float*)alloc(MD * 4);   // reused for both residual sums
  float*  hF     = (float*)alloc(MD * 4);
  __bf16* hbf    = (__bf16*)alloc(MD * 2);
  __bf16* ff1bf  = (__bf16*)alloc(MF * 2);

  // 1) Precision conversion / weight transposition.
  convert_bf16<<<(int)(MD / 256), 256, 0, stream>>>(x, xbf, (long)MD);
  transpose_convert_head<<<(DD * DD) / 256, 256, 0, stream>>>(Wq, WqT);
  transpose_convert_head<<<(DD * DD) / 256, 256, 0, stream>>>(Wk, WkT);
  transpose_convert_head<<<(DD * DD) / 256, 256, 0, stream>>>(Wv, WvT);
  transpose_convert<<<(DD * DD) / 256, 256, 0, stream>>>(Wo, WoT, DD, DD);
  transpose_convert<<<(DD * FF) / 256, 256, 0, stream>>>(W1, W1T, DD, FF);
  transpose_convert<<<(FF * DD) / 256, 256, 0, stream>>>(W2, W2T, FF, DD);

  // 2) QKV projections (bias fused, bf16 out).
  dim3 gD(DD / 128, MB / 128);
  gemm_bf16<0><<<gD, 128, 0, stream>>>(xbf, WqT, bq, nullptr, nullptr, qbf, MB, DD, DD);
  gemm_bf16<0><<<gD, 128, 0, stream>>>(xbf, WkT, bk, nullptr, nullptr, kbf, MB, DD, DD);
  gemm_bf16<0><<<gD, 128, 0, stream>>>(xbf, WvT, bv, nullptr, nullptr, vbf, MB, DD, DD);

  // 3) Attention (flash, online softmax, head-concatenated bf16 output).
  dim3 gA(SS / 64, HH, 4);
  flash_attn<<<gA, 128, 0, stream>>>(qbf, kbf, vbf, attnbf);

  // 4) Output projection + residual (x + attn@Wo + bo) -> f32.
  gemm_bf16<1><<<gD, 128, 0, stream>>>(attnbf, WoT, bo, x, resid, nullptr, MB, DD, DD);

  // 5) LayerNorm 1 -> hF (f32 for residual 2) and hbf (bf16 GEMM input).
  layernorm_k<<<MB, 256, 0, stream>>>(resid, ln1g, ln1b, hF, hbf, 1);

  // 6) FFN1 + exact GELU -> bf16.
  dim3 gF(FF / 128, MB / 128);
  gemm_bf16<2><<<gF, 128, 0, stream>>>(hbf, W1T, b1, nullptr, nullptr, ff1bf, MB, FF, DD);

  // 7) FFN2 + residual (h + ff@W2 + b2) -> f32.
  gemm_bf16<1><<<gD, 128, 0, stream>>>(ff1bf, W2T, b2, hF, resid, nullptr, MB, DD, FF);

  // 8) LayerNorm 2 -> final f32 output.
  layernorm_k<<<MB, 256, 0, stream>>>(resid, ln2g, ln2b, out, nullptr, 0);
}
